// PoleAgent004_37572373905494
// MI455X (gfx1250) — compile-verified, matrix operand order checked by reading, not run on hardware
//
#include <hip/hip_runtime.h>

// Problem constants
#define NY 64
#define NU 64
#define NH 512
#define H2 256
#define NB 256
#define T  1024
#define NCHUNK 16
#define CLEN   64   // T / NCHUNK

typedef __attribute__((ext_vector_type(16))) _Float16 v16h;
typedef __attribute__((ext_vector_type(8)))  float    v8f;
typedef __attribute__((ext_vector_type(4)))  float    v4f;
typedef __attribute__((ext_vector_type(4)))  _Float16 v4h;
typedef __attribute__((ext_vector_type(4)))  unsigned int v4u;

union H16 { v16h h; v4u u[2]; };

// ---------------- prep kernels ----------------

__global__ void k_prep_lam(const float* __restrict__ lr, const float* __restrict__ li,
                           float* __restrict__ lam_re, float* __restrict__ lam_im,
                           float* __restrict__ lamC_re, float* __restrict__ lamC_im,
                           float* __restrict__ nf) {
  int i = threadIdx.x;
  float e  = expf(-fabsf(lr[i]));
  float r  = expf(-e);
  float th = 1.5707963267948966f * expf(-fabsf(li[i]));
  lam_re[i] = r * cosf(th);
  lam_im[i] = r * sinf(th);
  nf[i] = sqrtf(fmaxf(0.f, 1.f - expf(-2.f * e)));
  float rL  = expf(-(float)CLEN * e);            // lam^CLEN
  float thL = (float)CLEN * th;
  lamC_re[i] = rL * cosf(thL);
  lamC_im[i] = rL * sinf(thL);
}

__global__ void k_prep_B(const float* __restrict__ B, const float* __restrict__ nf,
                         _Float16* __restrict__ Bw) {
  int i = blockIdx.x * blockDim.x + threadIdx.x;      // NH*NU
  if (i < NH * NU) {
    int h = i >> 6;
    Bw[i] = (_Float16)(B[i] * nf[h & (H2 - 1)]);
  }
}

__global__ void k_prep_W(const float* __restrict__ W, _Float16* __restrict__ Wh) {
  int i = blockIdx.x * blockDim.x + threadIdx.x;      // NY*NH
  if (i < NY * NH) Wh[i] = (_Float16)W[i];
}

__global__ void k_prep_U(const v4f* __restrict__ U, v4h* __restrict__ Uh, int n4) {
  int i = blockIdx.x * blockDim.x + threadIdx.x;
  if (i < n4) {
    v4f x = U[i];
    v4h o;
#pragma unroll
    for (int v = 0; v < 4; ++v) o[v] = (_Float16)x[v];
    Uh[i] = o;
  }
}

// x0[b][h] = y0 @ Wy2x^T + by2x
__global__ void k_x0(const float* __restrict__ y0, const float* __restrict__ Wy2x,
                     const float* __restrict__ by2x, float* __restrict__ x0) {
  int i = blockIdx.x * blockDim.x + threadIdx.x;      // NB*NH
  if (i >= NB * NH) return;
  int b = i >> 9, h = i & (NH - 1);
  const float* yr = y0 + b * NY;
  const float* wr = Wy2x + h * NY;
  float s = by2x[h];
#pragma unroll 8
  for (int k = 0; k < NY; ++k) s = fmaf(yr[k], wr[k], s);
  x0[i] = s;
}

// Y[0] = x0 @ Wx2y^T + bx2y  (tiny, keeps the scan loop branch-free)
__global__ void k_y0(const float* __restrict__ x0, const float* __restrict__ W,
                     const float* __restrict__ bx2y, float* __restrict__ Y) {
  int i = blockIdx.x * blockDim.x + threadIdx.x;      // NB*NY
  if (i >= NB * NY) return;
  int b = i >> 6, y = i & (NY - 1);
  const float* xr = x0 + (size_t)b * NH;
  const float* wr = W + (size_t)y * NH;
  float s = bx2y[y];
#pragma unroll 8
  for (int k = 0; k < NH; ++k) s = fmaf(xr[k], wr[k], s);
  Y[(size_t)b * NY + y] = s;
}

// ---------------- pass A: zero-state chunk endpoints ----------------
// grid (NB/16, NCHUNK), block 512 (16 waves). Wave w owns h2-slice [w*16,w*16+16).
// State tiles: M = h rows, N = batch cols.  U operand register double-buffered.
__global__ __launch_bounds__(512) void k_scanA(
    const _Float16* __restrict__ U16, const _Float16* __restrict__ Bw,
    const float* __restrict__ lam_re, const float* __restrict__ lam_im,
    const float* __restrict__ bias, float* __restrict__ Z) {
  const int mbt = blockIdx.x, c = blockIdx.y;
  const int tid = threadIdx.x;
  const int w = tid >> 5, lane = tid & 31, g = lane >> 4, r = lane & 15;
  const int mb = mbt * 16;
  const int hA = w * 16, hB = H2 + w * 16;

  v8f lre, lim, bA, bB;
#pragma unroll
  for (int v = 0; v < 8; ++v) {
    int h2 = hA + g * 8 + v;
    lre[v] = lam_re[h2]; lim[v] = lam_im[h2];
    bA[v] = bias[h2];    bB[v] = bias[H2 + h2];
  }

  H16 Bop[2][2];
#pragma unroll
  for (int j = 0; j < 2; ++j) {
    const _Float16* p = Bw + (size_t)((j ? hB : hA) + r) * NU;
#pragma unroll
    for (int kb = 0; kb < 2; ++kb) {
      Bop[j][kb].u[0] = *(const v4u*)(p + kb * 32 + g * 8);
      Bop[j][kb].u[1] = *(const v4u*)(p + kb * 32 + 16 + g * 8);
    }
  }

  v8f s0 = {}, s1 = {};
  const _Float16* Ub = U16 + ((size_t)(c * CLEN) * NB + mb + r) * NU;

  // prefetch t=0
  H16 Uc[2];
#pragma unroll
  for (int kb = 0; kb < 2; ++kb) {
    Uc[kb].u[0] = *(const v4u*)(Ub + kb * 32 + g * 8);
    Uc[kb].u[1] = *(const v4u*)(Ub + kb * 32 + 16 + g * 8);
  }

  for (int t = 0; t < CLEN; ++t) {
    v8f r0, r1;
#pragma unroll
    for (int v = 0; v < 8; ++v) {
      r0[v] = fmaf(lre[v], s0[v], fmaf(-lim[v], s1[v], bA[v]));
      r1[v] = fmaf(lim[v], s0[v], fmaf( lre[v], s1[v], bB[v]));
    }
    // prefetch t+1 (one row past the chunk end still lands inside the workspace)
    H16 Un[2];
    const _Float16* up = Ub + (size_t)(t + 1) * (NB * NU);
#pragma unroll
    for (int kb = 0; kb < 2; ++kb) {
      Un[kb].u[0] = *(const v4u*)(up + kb * 32 + g * 8);
      Un[kb].u[1] = *(const v4u*)(up + kb * 32 + 16 + g * 8);
    }
    s0 = __builtin_amdgcn_wmma_f32_16x16x32_f16(false, Bop[0][0].h, false, Uc[0].h, (short)0, r0, false, false);
    s0 = __builtin_amdgcn_wmma_f32_16x16x32_f16(false, Bop[0][1].h, false, Uc[1].h, (short)0, s0, false, false);
    s1 = __builtin_amdgcn_wmma_f32_16x16x32_f16(false, Bop[1][0].h, false, Uc[0].h, (short)0, r1, false, false);
    s1 = __builtin_amdgcn_wmma_f32_16x16x32_f16(false, Bop[1][1].h, false, Uc[1].h, (short)0, s1, false, false);
    Uc[0] = Un[0]; Uc[1] = Un[1];
  }

  float* z = Z + ((size_t)c * NB + mb + r) * NH;
#pragma unroll
  for (int v = 0; v < 8; ++v) {
    z[hA + g * 8 + v] = s0[v];
    z[hB + g * 8 + v] = s1[v];
  }
}

// ---------------- chunk combine: s_{c+1} = lam^CLEN ∘ s_c + z_c ----------------
__global__ void k_combine(const float* __restrict__ x0, const float* __restrict__ Z,
                          const float* __restrict__ lamC_re, const float* __restrict__ lamC_im,
                          float* __restrict__ S) {
  int i = blockIdx.x * blockDim.x + threadIdx.x;      // NB*H2
  if (i >= NB * H2) return;
  int b = i >> 8, h2 = i & (H2 - 1);
  float re = x0[(size_t)b * NH + h2];
  float im = x0[(size_t)b * NH + H2 + h2];
  float cr = lamC_re[h2], ci = lamC_im[h2];
  for (int c = 0; c < NCHUNK; ++c) {
    size_t o = ((size_t)c * NB + b) * NH;
    S[o + h2] = re;
    S[o + H2 + h2] = im;
    if (c + 1 < NCHUNK) {
      float zr = Z[o + h2], zi = Z[o + H2 + h2];
      float nr = fmaf(cr, re, fmaf(-ci, im, zr));
      float ni = fmaf(ci, re, fmaf( cr, im, zi));
      re = nr; im = ni;
    }
  }
}

// ---------------- pass C: scan from true start state, fused Y, async-LDS U ----------------
// The 2KB U tile per (workgroup, step) is shared by all 16 waves -> stage it once via
// global_load_async_to_lds_b128 (waves 0-3, one b128/lane), double-buffered 2 steps ahead,
// synchronized with s_wait_asynccnt + the barriers the Y reduction needs anyway.
__global__ __launch_bounds__(512) void k_scanC(
    const _Float16* __restrict__ U16, const _Float16* __restrict__ Bw,
    const _Float16* __restrict__ Wh,
    const float* __restrict__ lam_re, const float* __restrict__ lam_im,
    const float* __restrict__ bias, const float* __restrict__ S,
    const float* __restrict__ bx2y, float* __restrict__ Y) {
  __shared__ float yred[16 * 1088];                       // [wave][y][17] padded
  __shared__ __align__(16) _Float16 ubuf[2][16 * 72];     // 2 x (16 rows x 64, padded to 72)

  const int mbt = blockIdx.x, c = blockIdx.y;
  const int tid = threadIdx.x;
  const int w = tid >> 5, lane = tid & 31, g = lane >> 4, r = lane & 15;
  const int mb = mbt * 16;
  const int hA = w * 16, hB = H2 + w * 16;

  v8f lre, lim, bA, bB;
#pragma unroll
  for (int v = 0; v < 8; ++v) {
    int h2 = hA + g * 8 + v;
    lre[v] = lam_re[h2]; lim[v] = lam_im[h2];
    bA[v] = bias[h2];    bB[v] = bias[H2 + h2];
  }

  H16 Bop[2][2];
#pragma unroll
  for (int j = 0; j < 2; ++j) {
    const _Float16* p = Bw + (size_t)((j ? hB : hA) + r) * NU;
#pragma unroll
    for (int kb = 0; kb < 2; ++kb) {
      Bop[j][kb].u[0] = *(const v4u*)(p + kb * 32 + g * 8);
      Bop[j][kb].u[1] = *(const v4u*)(p + kb * 32 + 16 + g * 8);
    }
  }
  H16 Wop[4];
#pragma unroll
  for (int j = 0; j < 4; ++j) {
    const _Float16* p = Wh + (size_t)(j * 16 + r) * NH;
    Wop[j].u[0] = *(const v4u*)(p + hA + g * 8);
    Wop[j].u[1] = *(const v4u*)(p + hB + g * 8);
  }

  const float* sc = S + ((size_t)c * NB + mb + r) * NH;
  v8f s0, s1;
#pragma unroll
  for (int v = 0; v < 8; ++v) {
    s0[v] = sc[hA + g * 8 + v];
    s1[v] = sc[hB + g * 8 + v];
  }

  // async U staging state (tile = contiguous 2KB of U16; lanes of waves 0-3 cover it)
  const char* ugbase = (const char*)(U16 + ((size_t)(c * CLEN) * NB + mb) * NU);
  const unsigned uli  = (unsigned)(w * 32 + lane);        // 0..127 for w<4
  const unsigned lrow = uli >> 3, lcol = uli & 7;         // 8 x 16B per 128B row
  const unsigned ldsA = (unsigned)(uintptr_t)(&ubuf[0][0]) + lrow * 144u + lcol * 16u;
  const unsigned ldsB = (unsigned)(uintptr_t)(&ubuf[1][0]) + lrow * 144u + lcol * 16u;

  auto issueU = [&](int t, unsigned ldsa) {
    const char* ga = ugbase + (size_t)t * (NB * NU * 2) + (size_t)uli * 16;
    asm volatile("global_load_async_to_lds_b128 %0, %1, off"
                 :: "v"(ldsa), "v"(ga) : "memory");
  };
  if (w < 4) { issueU(0, ldsA); issueU(1, ldsB); }

  for (int t = 0; t < CLEN; ++t) {
    // issuing waves: all but the newest async load (t+1's) must be complete
    asm volatile("s_wait_asynccnt 0x1" ::: "memory");
    __syncthreads();           // publish ubuf[t&1]; yred from step t-1 fully read

    const int p = t & 1;
    v8f r0, r1;
#pragma unroll
    for (int v = 0; v < 8; ++v) {
      r0[v] = fmaf(lre[v], s0[v], fmaf(-lim[v], s1[v], bA[v]));
      r1[v] = fmaf(lim[v], s0[v], fmaf( lre[v], s1[v], bB[v]));
    }
    H16 Uop[2];
    const _Float16* ub = &ubuf[p][0] + (size_t)r * 72;
#pragma unroll
    for (int kb = 0; kb < 2; ++kb) {
      Uop[kb].u[0] = *(const v4u*)(ub + kb * 32 + g * 8);
      Uop[kb].u[1] = *(const v4u*)(ub + kb * 32 + 16 + g * 8);
    }
    s0 = __builtin_amdgcn_wmma_f32_16x16x32_f16(false, Bop[0][0].h, false, Uop[0].h, (short)0, r0, false, false);
    s0 = __builtin_amdgcn_wmma_f32_16x16x32_f16(false, Bop[0][1].h, false, Uop[1].h, (short)0, s0, false, false);
    s1 = __builtin_amdgcn_wmma_f32_16x16x32_f16(false, Bop[1][0].h, false, Uop[0].h, (short)0, r1, false, false);
    s1 = __builtin_amdgcn_wmma_f32_16x16x32_f16(false, Bop[1][1].h, false, Uop[1].h, (short)0, s1, false, false);

    // Y partials: state C-layout matches WMMA K-striping -> per-lane cvt only
    H16 xop;
#pragma unroll
    for (int v = 0; v < 8; ++v) {
      xop.h[v]     = (_Float16)s0[v];
      xop.h[8 + v] = (_Float16)s1[v];
    }
    v8f acc[4];
#pragma unroll
    for (int j = 0; j < 4; ++j) {
      v8f zero = {};
      acc[j] = __builtin_amdgcn_wmma_f32_16x16x32_f16(false, Wop[j].h, false, xop.h, (short)0, zero, false, false);
    }
#pragma unroll
    for (int j = 0; j < 4; ++j)
#pragma unroll
      for (int v = 0; v < 8; ++v)
        yred[w * 1088 + (j * 16 + g * 8 + v) * 17 + r] = acc[j][v];
    __syncthreads();           // partials visible; everyone past reading ubuf[p]

    // overwrite ubuf[p] with the tile for step t+2 (stays inside workspace past end)
    if (w < 4) issueU(t + 2, p ? ldsB : ldsA);

    // K-split reduction over 16 waves; coalesced Y store (consecutive tid -> consecutive y)
    const int tg = c * CLEN + t + 1;
#pragma unroll
    for (int o = 0; o < 2; ++o) {
      int f = o * 512 + tid;
      int y = f & 63, b2 = f >> 6;
      float sum = 0.f;
#pragma unroll
      for (int ww = 0; ww < 16; ++ww) sum += yred[ww * 1088 + y * 17 + b2];
      Y[((size_t)tg * NB + mb + b2) * NY + y] = sum + bx2y[y];
    }
  }
}

// ---------------- host launcher ----------------
extern "C" void kernel_launch(void* const* d_in, const int* in_sizes, int n_in,
                              void* d_out, int out_size, void* d_ws, size_t ws_size,
                              hipStream_t stream) {
  (void)in_sizes; (void)n_in; (void)out_size; (void)ws_size;
  const float* y0   = (const float*)d_in[0];
  const float* U    = (const float*)d_in[1];
  const float* llr  = (const float*)d_in[2];
  const float* lli  = (const float*)d_in[3];
  const float* B    = (const float*)d_in[4];
  const float* bias = (const float*)d_in[5];
  const float* Wy2x = (const float*)d_in[6];
  const float* by2x = (const float*)d_in[7];
  const float* Wx2y = (const float*)d_in[8];
  const float* bx2y = (const float*)d_in[9];
  float* Y = (float*)d_out;

  char* ws = (char*)d_ws;
  float*    lam_re  = (float*)   (ws + 0x0);
  float*    lam_im  = (float*)   (ws + 0x400);
  float*    lamC_re = (float*)   (ws + 0x800);
  float*    lamC_im = (float*)   (ws + 0xC00);
  float*    nf      = (float*)   (ws + 0x1000);
  _Float16* Bw      = (_Float16*)(ws + 0x2000);     //  64 KB
  _Float16* Wh      = (_Float16*)(ws + 0x12000);    //  64 KB
  float*    x0      = (float*)   (ws + 0x22000);    // 512 KB
  _Float16* U16     = (_Float16*)(ws + 0x100000);   //  32 MB
  float*    Z       = (float*)   (ws + 0x2100000);  //   8 MB
  float*    S       = (float*)   (ws + 0x2900000);  //   8 MB (ends ~51.4 MB)

  k_prep_lam<<<1, 256, 0, stream>>>(llr, lli, lam_re, lam_im, lamC_re, lamC_im, nf);
  k_prep_B  <<<(NH * NU + 255) / 256, 256, 0, stream>>>(B, nf, Bw);
  k_prep_W  <<<(NY * NH + 255) / 256, 256, 0, stream>>>(Wx2y, Wh);
  k_prep_U  <<<(T * NB * NU / 4 + 255) / 256, 256, 0, stream>>>((const v4f*)U, (v4h*)U16, T * NB * NU / 4);
  k_x0      <<<(NB * NH + 255) / 256, 256, 0, stream>>>(y0, Wy2x, by2x, x0);
  k_y0      <<<(NB * NY + 255) / 256, 256, 0, stream>>>(x0, Wx2y, bx2y, Y);

  k_scanA   <<<dim3(NB / 16, NCHUNK), 512, 0, stream>>>(U16, Bw, lam_re, lam_im, bias, Z);
  k_combine <<<(NB * H2 + 255) / 256, 256, 0, stream>>>(x0, Z, lamC_re, lamC_im, S);
  k_scanC   <<<dim3(NB / 16, NCHUNK), 512, 0, stream>>>(U16, Bw, Wh, lam_re, lam_im, bias, S, bx2y, Y);
}
